// DHT_Layer_45002667327995
// MI455X (gfx1250) — matrix operand order repaired and checked by reading, not compiled, not run on hardware
//
#include <hip/hip_runtime.h>
#include <math.h>

// Problem constants
#define BQ   4
#define CINQ 128
#define DIMC 128
#define Hh   128
#define Ww   128
#define HWQ  (Hh*Ww)          // 16384
#define NA   180
#define NR   184
#define NRP  192              // padded histogram row
#define REP  8                // LDS histogram replicas
#define BN_RSQ 0.9999950000374997f   // 1/sqrt(1+1e-5)

typedef __attribute__((ext_vector_type(16))) __bf16 v16bf;
typedef __attribute__((ext_vector_type(8)))  float  v8f;

union FragBF { v16bf v; unsigned u[8]; };

__device__ __forceinline__ unsigned short f2bf(float f) {
    union { float f; unsigned u; } x; x.f = f;
    unsigned r = x.u + 0x7FFFu + ((x.u >> 16) & 1u);   // round-to-nearest-even
    return (unsigned short)(r >> 16);
}

// ---------------- packing kernels ----------------

// x [B][CIN][HW] f32  ->  xT [B][HW][CIN] bf16
__global__ void pack_xT_kernel(const float* __restrict__ x, unsigned short* __restrict__ xT) {
    size_t idx = (size_t)blockIdx.x * blockDim.x + threadIdx.x;   // B*HW*CIN threads, ci fastest
    int ci = (int)(idx & (CINQ - 1));
    size_t rest = idx >> 7;
    int p = (int)(rest & (HWQ - 1));
    int b = (int)(rest >> 14);
    xT[idx] = f2bf(x[((size_t)b * CINQ + ci) * HWQ + p]);
}

// w1 [DIM][CIN] f32 -> bf16 (same layout)
__global__ void pack_w1_kernel(const float* __restrict__ w, unsigned short* __restrict__ wp) {
    int idx = blockIdx.x * blockDim.x + threadIdx.x;
    if (idx < DIMC * CINQ) wp[idx] = f2bf(w[idx]);
}

// w [DIM][CIN][3][3] f32 -> wp [tap][DIM][CIN] bf16
__global__ void pack_w3x3_kernel(const float* __restrict__ w, unsigned short* __restrict__ wp) {
    int idx = blockIdx.x * blockDim.x + threadIdx.x;   // 9*DIM*CIN, ci fastest
    if (idx >= 9 * DIMC * CINQ) return;
    int ci   = idx & (CINQ - 1);
    int rest = idx >> 7;
    int co   = rest & (DIMC - 1);
    int tap  = rest >> 7;
    wp[idx] = f2bf(w[((size_t)co * CINQ + ci) * 9 + tap]);
}

// rho-bin index table: ridx[a][p] (uint8), double math + rint to match np.round
__global__ void ridx_kernel(unsigned char* __restrict__ ridx) {
    int idx = blockIdx.x * blockDim.x + threadIdx.x;
    if (idx >= NA * HWQ) return;
    int a = idx / HWQ, p = idx - a * HWQ;
    int i = p >> 7, j = p & 127;
    const double PI_D = 3.14159265358979323846;
    double theta = (double)a * (PI_D / (double)NA);
    double xc = (double)(j - Ww / 2);
    double yc = (double)(i - Hh / 2);
    double irho = sqrt((double)(Hh * Hh + Ww * Ww)) / (double)NR;
    double r = rint((xc * cos(theta) + yc * sin(theta)) / irho) + (double)(NR / 2);
    int ri = (int)r;
    ri = ri < 0 ? 0 : (ri > NR - 1 ? NR - 1 : ri);
    ridx[idx] = (unsigned char)ri;
}

// ---------------- 1x1 conv + BN + ReLU (WMMA GEMM) ----------------
// All 8 co-tiles accumulated in registers; B fragment loaded once per K-tile.
__global__ __launch_bounds__(256) void conv1x1_wmma(
    const unsigned short* __restrict__ xT,   // [B][HW][CIN] bf16
    const unsigned short* __restrict__ w1p,  // [DIM][CIN]   bf16
    const float* __restrict__ b1, const float* __restrict__ g1, const float* __restrict__ be1,
    float* __restrict__ x1) {                // [B][DIM][HW] f32
    const int lane = threadIdx.x & 31;
    const int half = lane >> 4;
    const int mn   = lane & 15;
    const int gw = blockIdx.x * (blockDim.x >> 5) + (threadIdx.x >> 5);  // 4096 waves
    const int b  = gw >> 10;
    const int p0 = (gw & 1023) << 4;
    const unsigned short* brow = xT + ((size_t)b * HWQ + p0 + mn) * CINQ;

    v8f acc[8];
    #pragma unroll
    for (int ct = 0; ct < 8; ++ct) acc[ct] = (v8f){0.f,0.f,0.f,0.f,0.f,0.f,0.f,0.f};

    #pragma unroll
    for (int kt = 0; kt < 4; ++kt) {
        FragBF Bf;
        #pragma unroll
        for (int v = 0; v < 8; ++v)
            Bf.u[v] = *(const unsigned*)(brow + kt * 32 + ((v < 4) ? 2 * v : 16 + 2 * (v - 4)) + 8 * half);
        #pragma unroll
        for (int ct = 0; ct < 8; ++ct) {
            const unsigned short* arow = w1p + (size_t)(ct * 16 + mn) * CINQ + kt * 32;
            FragBF A;
            #pragma unroll
            for (int v = 0; v < 8; ++v) {
                int ka = ((v < 4) ? 2 * v : 16 + 2 * (v - 4)) + 8 * half;
                A.u[v] = *(const unsigned*)(arow + ka);
            }
            acc[ct] = __builtin_amdgcn_wmma_f32_16x16x32_bf16(false, A.v, false, Bf.v,
                                                              (short)0, acc[ct], false, false);
        }
    }
    #pragma unroll
    for (int ct = 0; ct < 8; ++ct) {
        #pragma unroll
        for (int v = 0; v < 8; ++v) {
            int co = ct * 16 + v + 8 * half;
            float y = acc[ct][v] + b1[co];
            y = y * (g1[co] * BN_RSQ) + be1[co];
            y = fmaxf(y, 0.f);
            x1[((size_t)b * DIMC + co) * HWQ + p0 + mn] = y;
        }
    }
}

// ---------------- DHT scatter + back-projection + sigmoid mask ----------------
// One block per (b,c). 8-way replicated LDS histogram cuts ds_add_f32 collisions.
__global__ __launch_bounds__(256) void dht_kernel(
    const float* __restrict__ x1,            // [B*DIM][HW]
    const unsigned char* __restrict__ ridx,  // [A][HW]
    float* __restrict__ mask) {              // [B*DIM][HW]
    __shared__ float hist[REP][NRP];
    __shared__ float hsum[NRP];
    const int tid = threadIdx.x;
    const int rep = tid & (REP - 1);
    const int bc  = blockIdx.x;
    const float* xs = x1 + (size_t)bc * HWQ;
    float xv[64], bs[64];
    #pragma unroll
    for (int i = 0; i < 64; ++i) { xv[i] = xs[tid + (i << 8)]; bs[i] = 0.f; }
    for (int a = 0; a < NA; ++a) {
        #pragma unroll
        for (int z = tid; z < REP * NRP; z += 256) ((float*)hist)[z] = 0.f;
        __syncthreads();
        const unsigned char* rr = ridx + (size_t)a * HWQ;
        #pragma unroll 8
        for (int i = 0; i < 64; ++i)
            atomicAdd(&hist[rep][rr[tid + (i << 8)]], xv[i]);   // ds_add_f32
        __syncthreads();
        if (tid < NR) {
            float s = 0.f;
            #pragma unroll
            for (int r = 0; r < REP; ++r) s += hist[r][tid];
            hsum[tid] = s;
        }
        __syncthreads();
        #pragma unroll 8
        for (int i = 0; i < 64; ++i)
            bs[i] += hsum[rr[tid + (i << 8)]];
        __syncthreads();
    }
    float* mrow = mask + (size_t)bc * HWQ;
    #pragma unroll
    for (int i = 0; i < 64; ++i)
        mrow[tid + (i << 8)] = 1.f / (1.f + expf(-bs[i] * (1.f / (float)NA)));
}

// ---------------- 3x3 conv + BN + ReLU (implicit-GEMM WMMA) ----------------
// 8 co-tile accumulators in registers; each B fragment feeds 8 WMMAs.
// FUSE=false: write bf16 transposed activation [B][HW][DIM] for the next conv.
// FUSE=true : write f32 NCHW output gated by mask.
template <bool FUSE>
__global__ __launch_bounds__(256) void conv3x3_wmma(
    const unsigned short* __restrict__ inT,   // [B][HW][K=128] bf16
    const unsigned short* __restrict__ wp,    // [9][DIM][K]    bf16
    const float* __restrict__ bias, const float* __restrict__ g, const float* __restrict__ be,
    unsigned short* __restrict__ outT,        // used when !FUSE
    const float* __restrict__ mask,           // used when  FUSE
    float* __restrict__ out) {                // used when  FUSE
    const int lane = threadIdx.x & 31;
    const int half = lane >> 4;
    const int mn   = lane & 15;
    const int gw = blockIdx.x * (blockDim.x >> 5) + (threadIdx.x >> 5);
    const int b   = gw >> 10;
    const int p0  = (gw & 1023) << 4;          // 16 pixels, same row
    const int row = p0 >> 7;
    const int col = (p0 & 127) + mn;

    v8f acc[8];
    #pragma unroll
    for (int ct = 0; ct < 8; ++ct) acc[ct] = (v8f){0.f,0.f,0.f,0.f,0.f,0.f,0.f,0.f};

    for (int tap = 0; tap < 9; ++tap) {
        const int ky = tap / 3, kx = tap - 3 * ky;
        const int sy = row + ky - 1;
        const int sx = col + kx - 1;
        const bool valid = (sy >= 0) && (sy < Hh) && (sx >= 0) && (sx < Ww);
        const long sp = valid ? ((long)b * HWQ + (long)sy * Ww + sx) : 0;
        const unsigned short* brow = inT + (size_t)sp * DIMC;
        const unsigned short* wtap = wp + (size_t)tap * DIMC * DIMC;
        __builtin_prefetch(brow, 0, 0);
        #pragma unroll
        for (int kt = 0; kt < 4; ++kt) {
            FragBF Bf;
            #pragma unroll
            for (int v = 0; v < 8; ++v) {
                unsigned raw = *(const unsigned*)(brow + kt * 32 + ((v < 4) ? 2 * v : 16 + 2 * (v - 4)) + 8 * half);
                Bf.u[v] = valid ? raw : 0u;     // branchless: EXEC stays full for WMMA
            }
            #pragma unroll
            for (int ct = 0; ct < 8; ++ct) {
                const unsigned short* arow = wtap + (size_t)(ct * 16 + mn) * DIMC + kt * 32;
                FragBF A;
                #pragma unroll
                for (int v = 0; v < 8; ++v) {
                    int ka = ((v < 4) ? 2 * v : 16 + 2 * (v - 4)) + 8 * half;
                    A.u[v] = *(const unsigned*)(arow + ka);
                }
                acc[ct] = __builtin_amdgcn_wmma_f32_16x16x32_bf16(false, A.v, false, Bf.v,
                                                                  (short)0, acc[ct], false, false);
            }
        }
    }

    #pragma unroll
    for (int ct = 0; ct < 8; ++ct) {
        const int cb = ct * 16 + 8 * half;
        float y[8];
        #pragma unroll
        for (int v = 0; v < 8; ++v) {
            int co = cb + v;
            float t = acc[ct][v] + bias[co];
            t = t * (g[co] * BN_RSQ) + be[co];
            y[v] = fmaxf(t, 0.f);
        }
        if (FUSE) {
            #pragma unroll
            for (int v = 0; v < 8; ++v) {
                int co = cb + v;
                size_t o = ((size_t)b * DIMC + co) * HWQ + p0 + mn;
                out[o] = y[v] * mask[o];
            }
        } else {
            unsigned pk[4];
            #pragma unroll
            for (int j = 0; j < 4; ++j)
                pk[j] = (unsigned)f2bf(y[2 * j]) | ((unsigned)f2bf(y[2 * j + 1]) << 16);
            uint4 val; val.x = pk[0]; val.y = pk[1]; val.z = pk[2]; val.w = pk[3];
            *(uint4*)(outT + ((size_t)b * HWQ + p0 + mn) * DIMC + cb) = val;  // 16B aligned
        }
    }
}

// ---------------- launch ----------------
extern "C" void kernel_launch(void* const* d_in, const int* in_sizes, int n_in,
                              void* d_out, int out_size, void* d_ws, size_t ws_size,
                              hipStream_t stream) {
    (void)in_sizes; (void)n_in; (void)out_size; (void)ws_size;
    const float* x   = (const float*)d_in[0];
    const float* w1  = (const float*)d_in[1];
    const float* b1  = (const float*)d_in[2];
    const float* g1  = (const float*)d_in[3];
    const float* be1 = (const float*)d_in[4];
    const float* w2  = (const float*)d_in[5];
    const float* b2  = (const float*)d_in[6];
    const float* g2  = (const float*)d_in[7];
    const float* be2 = (const float*)d_in[8];
    const float* w3  = (const float*)d_in[9];
    const float* b3  = (const float*)d_in[10];
    const float* g3  = (const float*)d_in[11];
    const float* be3 = (const float*)d_in[12];

    char* ws = (char*)d_ws;
    size_t off = 0;
    auto wsalloc = [&](size_t bytes) -> char* {
        char* p = ws + off;
        off = (off + bytes + 255) & ~(size_t)255;
        return p;
    };
    unsigned short* xT   = (unsigned short*)wsalloc((size_t)BQ * HWQ * CINQ * 2);
    unsigned short* w1p  = (unsigned short*)wsalloc((size_t)DIMC * CINQ * 2);
    unsigned short* w2p  = (unsigned short*)wsalloc((size_t)9 * DIMC * CINQ * 2);
    unsigned short* w3p  = (unsigned short*)wsalloc((size_t)9 * DIMC * CINQ * 2);
    unsigned short* h1T  = (unsigned short*)wsalloc((size_t)BQ * HWQ * DIMC * 2);
    float*          x1   = (float*)wsalloc((size_t)BQ * DIMC * HWQ * 4);
    float*          mask = (float*)wsalloc((size_t)BQ * DIMC * HWQ * 4);
    unsigned char*  ridx = (unsigned char*)wsalloc((size_t)NA * HWQ);

    pack_xT_kernel<<<(BQ * HWQ * CINQ) / 256, 256, 0, stream>>>(x, xT);
    pack_w1_kernel<<<(DIMC * CINQ) / 256, 256, 0, stream>>>(w1, w1p);
    pack_w3x3_kernel<<<(9 * DIMC * CINQ + 255) / 256, 256, 0, stream>>>(w2, w2p);
    pack_w3x3_kernel<<<(9 * DIMC * CINQ + 255) / 256, 256, 0, stream>>>(w3, w3p);
    ridx_kernel<<<(NA * HWQ) / 256, 256, 0, stream>>>(ridx);

    conv1x1_wmma<<<512, 256, 0, stream>>>(xT, w1p, b1, g1, be1, x1);
    dht_kernel<<<BQ * DIMC, 256, 0, stream>>>(x1, ridx, mask);
    conv3x3_wmma<false><<<512, 256, 0, stream>>>(xT,  w2p, b2, g2, be2, h1T, nullptr, nullptr);
    conv3x3_wmma<true ><<<512, 256, 0, stream>>>(h1T, w3p, b3, g3, be3, nullptr, mask, (float*)d_out);
}
